// SNNMnist_13615046328450
// MI455X (gfx1250) — matrix operand order repaired and checked by reading, not compiled, not run on hardware
//
#include <hip/hip_runtime.h>
#include <hip/hip_bf16.h>

// ---------------------------------------------------------------------------
// SNN (norse LIF) forward on gfx1250.
//  - xproj = x @ W_in^T hoisted to one big bf16 WMMA GEMM (5120x4096x800pad)
//  - per-step fused kernel: rec = z_prev @ W_rec^T (bf16 WMMA, f32 accum)
//    + LIF pointwise update + spike gen + zsum accumulation
//  - readout hoisted: out = (zsum @ W_ro^T)/T + bias
//  Each wave computes a 2x2 block of 16x16 WMMA tiles: 8 b128 loads feed
//  4 v_wmma_f32_16x16x32_bf16 (2 loads/WMMA, each fragment reused twice).
// ---------------------------------------------------------------------------

#define T_STEPS 20
#define BATCH   256
#define IN_F    784
#define IN_PAD  800   // pad K to multiple of 32 for 16x16x32 WMMA
#define HID     4096
#define LAB     10

typedef __attribute__((ext_vector_type(16))) __bf16 bfrag;   // WMMA A/B operand (8 VGPRs)
typedef __attribute__((ext_vector_type(8)))  float  accfrag; // WMMA C/D operand (8 VGPRs)

struct Pair128 { uint4 lo, hi; };

// A-operand 16x32 bf16 from row-major source (row stride ldk).
// ISA layout: lane(l0,l1): elems 0..7 -> K=8*l1+j ; elems 8..15 -> K=16+8*l1+j ; M=l0.
__device__ __forceinline__ bfrag load_frag_a(const __hip_bfloat16* base, int row, int ldk,
                                             int k0, int l1) {
  const __hip_bfloat16* p = base + (size_t)row * ldk + k0 + 8 * l1;
  Pair128 t;
  t.lo = *reinterpret_cast<const uint4*>(p);
  t.hi = *reinterpret_cast<const uint4*>(p + 16);
  return __builtin_bit_cast(bfrag, t);
}

// B-operand 32x16 bf16: D = A * W^T, so B[k][n] = W[n][k]; each lane reads 16
// bf16 of one W row. ISA layout: lane(l0,l1): elems 0..7 -> K=8*l1+j ; elems 8..15 -> K=16+8*l1+j ; N=l0.
__device__ __forceinline__ bfrag load_frag_b(const __hip_bfloat16* base, int row, int ldk,
                                             int k0, int l1) {
  const __hip_bfloat16* p = base + (size_t)row * ldk + k0 + 8 * l1;
  Pair128 t;
  t.lo = *reinterpret_cast<const uint4*>(p);
  t.hi = *reinterpret_cast<const uint4*>(p + 16);
  return __builtin_bit_cast(bfrag, t);
}

__device__ __forceinline__ accfrag wmma_bf16(bfrag a, bfrag b, accfrag c) {
  return __builtin_amdgcn_wmma_f32_16x16x32_bf16(false, a, false, b, (short)0, c,
                                                 false, false);
}

// ---------------------------------------------------------------------------
// fp32 -> bf16 cast with optional K padding (zeros beyond kin).
__global__ void cast_pad_bf16_kernel(const float* __restrict__ in,
                                     __hip_bfloat16* __restrict__ out,
                                     int kin, int kout, int n) {
  int idx = blockIdx.x * blockDim.x + threadIdx.x;
  int stride = gridDim.x * blockDim.x;
  for (; idx < n; idx += stride) {
    int col = idx % kout;
    int row = idx / kout;
    float v = (col < kin) ? in[(size_t)row * kin + col] : 0.0f;
    out[idx] = __float2bfloat16(v);
  }
}

__global__ void zero_u32_kernel(unsigned int* __restrict__ p, int n) {
  int idx = blockIdx.x * blockDim.x + threadIdx.x;
  int stride = gridDim.x * blockDim.x;
  for (; idx < n; idx += stride) p[idx] = 0u;
}

// ---------------------------------------------------------------------------
// xproj[m][n] = sum_k x_pad[m][k] * w_in_pad[n][k]   (M=5120, N=4096, K=800)
// 128 threads = 4 waves; 64x64 WG tile; wave w owns 32x32 quadrant (w&1, w>>1).
__global__ void __launch_bounds__(128)
gemm_xproj_kernel(const __hip_bfloat16* __restrict__ A,   // [T*B][IN_PAD]
                  const __hip_bfloat16* __restrict__ B,   // [HID][IN_PAD]
                  float* __restrict__ D) {                // [T*B][HID]
  const int lane = threadIdx.x & 31;
  const int wv   = threadIdx.x >> 5;
  const int l0 = lane & 15, l1 = lane >> 4;
  const int mBase = blockIdx.x * 64 + (wv & 1) * 32;
  const int nBase = blockIdx.y * 64 + (wv >> 1) * 32;
  accfrag c00 = {}, c01 = {}, c10 = {}, c11 = {};
  const int aRow0 = mBase + l0;
  const int aRow1 = mBase + 16 + l0;
  const int bRow0 = nBase + l0;
  const int bRow1 = nBase + 16 + l0;
#pragma unroll 5
  for (int k0 = 0; k0 < IN_PAD; k0 += 32) {
    bfrag a0 = load_frag_a(A, aRow0, IN_PAD, k0, l1);
    bfrag a1 = load_frag_a(A, aRow1, IN_PAD, k0, l1);
    bfrag b0 = load_frag_b(B, bRow0, IN_PAD, k0, l1);
    bfrag b1 = load_frag_b(B, bRow1, IN_PAD, k0, l1);
    c00 = wmma_bf16(a0, b0, c00);
    c01 = wmma_bf16(a0, b1, c01);
    c10 = wmma_bf16(a1, b0, c10);
    c11 = wmma_bf16(a1, b1, c11);
  }
#pragma unroll
  for (int r = 0; r < 8; ++r) {
    int m0 = mBase + r + 8 * l1;
    int m1 = m0 + 16;
    D[(size_t)m0 * HID + nBase + l0]      = c00[r];
    D[(size_t)m0 * HID + nBase + 16 + l0] = c01[r];
    D[(size_t)m1 * HID + nBase + l0]      = c10[r];
    D[(size_t)m1 * HID + nBase + 16 + l0] = c11[r];
  }
}

// ---------------------------------------------------------------------------
// Fused per-timestep kernel:
//   rec = z_prev @ W_rec^T (this WG's 64x64 tile, full K=4096 reduction)
//   then LIF: v_dec = v + 0.05*(i - v); z = v_dec > 0.5; v' = (1-z)*v_dec;
//             i' = 0.9*i + xproj_t + rec;  zsum += z;  z stored bf16 (exact).
__global__ void __launch_bounds__(128)
lif_step_kernel(const __hip_bfloat16* __restrict__ zprev,  // [B][HID]
                __hip_bfloat16* __restrict__ zcur,         // [B][HID]
                const __hip_bfloat16* __restrict__ wrec,   // [HID][HID]
                const float* __restrict__ xproj_t,         // [B][HID] (t slice)
                float* __restrict__ vmem,
                float* __restrict__ isyn,
                float* __restrict__ zsum) {
  const int lane = threadIdx.x & 31;
  const int wv   = threadIdx.x >> 5;
  const int l0 = lane & 15, l1 = lane >> 4;
  const int mBase = blockIdx.x * 64 + (wv & 1) * 32;   // batch tile
  const int nBase = blockIdx.y * 64 + (wv >> 1) * 32;  // hidden tile
  accfrag c00 = {}, c01 = {}, c10 = {}, c11 = {};
  const int aRow0 = mBase + l0;
  const int aRow1 = mBase + 16 + l0;
  const int bRow0 = nBase + l0;
  const int bRow1 = nBase + 16 + l0;
#pragma unroll 2
  for (int k0 = 0; k0 < HID; k0 += 32) {
    bfrag a0 = load_frag_a(zprev, aRow0, HID, k0, l1);
    bfrag a1 = load_frag_a(zprev, aRow1, HID, k0, l1);
    bfrag b0 = load_frag_b(wrec, bRow0, HID, k0, l1);
    bfrag b1 = load_frag_b(wrec, bRow1, HID, k0, l1);
    c00 = wmma_bf16(a0, b0, c00);
    c01 = wmma_bf16(a0, b1, c01);
    c10 = wmma_bf16(a1, b0, c10);
    c11 = wmma_bf16(a1, b1, c11);
  }
#pragma unroll
  for (int mt = 0; mt < 2; ++mt) {
#pragma unroll
    for (int r = 0; r < 8; ++r) {
      int m = mBase + mt * 16 + r + 8 * l1;
#pragma unroll
      for (int nt = 0; nt < 2; ++nt) {
        int h = nBase + nt * 16 + l0;
        size_t idx = (size_t)m * HID + h;
        float rec = (mt == 0) ? ((nt == 0) ? c00[r] : c01[r])
                              : ((nt == 0) ? c10[r] : c11[r]);
        float v = vmem[idx];
        float i = isyn[idx];
        float vdec = v + 0.05f * (i - v);             // DT*TAU_MEM_INV = 0.05
        float z = (vdec > 0.5f) ? 1.0f : 0.0f;        // heaviside(v_dec - V_TH)
        vmem[idx] = vdec * (1.0f - z);                // reset to 0 on spike
        isyn[idx] = 0.9f * i + xproj_t[idx] + rec;    // i_dec + inputs + recurrent
        zcur[idx] = __float2bfloat16(z);              // 0.0 / 1.0 exact in bf16
        zsum[idx] += z;
      }
    }
  }
}

// ---------------------------------------------------------------------------
// out[b][l] = (sum_h zsum[b][h] * Wro[l][h]) / T + bias[l]
__global__ void __launch_bounds__(256)
readout_kernel(const float* __restrict__ zsum, const float* __restrict__ wro,
               const float* __restrict__ bias, float* __restrict__ out) {
  __shared__ float red[256];
  const int b = blockIdx.x;
  const int tid = threadIdx.x;
  float part[LAB];
#pragma unroll
  for (int l = 0; l < LAB; ++l) part[l] = 0.0f;
  for (int h = tid; h < HID; h += 256) {
    float s = zsum[(size_t)b * HID + h];
#pragma unroll
    for (int l = 0; l < LAB; ++l) part[l] += s * wro[(size_t)l * HID + h];
  }
  for (int l = 0; l < LAB; ++l) {
    red[tid] = part[l];
    __syncthreads();
    for (int s = 128; s > 0; s >>= 1) {
      if (tid < s) red[tid] += red[tid + s];
      __syncthreads();
    }
    if (tid == 0) out[b * LAB + l] = red[0] * (1.0f / T_STEPS) + bias[l];
    __syncthreads();
  }
}

// ---------------------------------------------------------------------------
extern "C" void kernel_launch(void* const* d_in, const int* in_sizes, int n_in,
                              void* d_out, int out_size, void* d_ws, size_t ws_size,
                              hipStream_t stream) {
  (void)in_sizes; (void)n_in; (void)out_size; (void)ws_size;
  const float* x    = (const float*)d_in[0];  // [T][B][IN_F]
  const float* win  = (const float*)d_in[1];  // [HID][IN_F]
  const float* wrec = (const float*)d_in[2];  // [HID][HID]
  const float* wro  = (const float*)d_in[3];  // [LAB][HID]
  const float* bias = (const float*)d_in[4];  // [LAB]
  float* out = (float*)d_out;                 // [B][LAB]

  // Workspace carve-out (all sizes multiples of 256B -> no padding gaps).
  char* ws = (char*)d_ws;
  size_t off = 0;
  auto carve = [&](size_t bytes) {
    void* p = ws + off;
    off += (bytes + 255) & ~(size_t)255;
    return p;
  };
  __hip_bfloat16* wrec_bf = (__hip_bfloat16*)carve((size_t)HID * HID * 2);        // 33.6 MB
  __hip_bfloat16* win_bf  = (__hip_bfloat16*)carve((size_t)HID * IN_PAD * 2);     //  6.6 MB
  __hip_bfloat16* x_bf    = (__hip_bfloat16*)carve((size_t)T_STEPS * BATCH * IN_PAD * 2); // 8.2 MB
  float* xproj = (float*)carve((size_t)T_STEPS * BATCH * HID * 4);                // 83.9 MB
  float* vmem  = (float*)carve((size_t)BATCH * HID * 4);                          //  4.2 MB
  float* isyn  = (float*)carve((size_t)BATCH * HID * 4);                          //  4.2 MB
  float* zsum  = (float*)carve((size_t)BATCH * HID * 4);                          //  4.2 MB
  __hip_bfloat16* z0 = (__hip_bfloat16*)carve((size_t)BATCH * HID * 2);           //  2.1 MB
  __hip_bfloat16* z1 = (__hip_bfloat16*)carve((size_t)BATCH * HID * 2);           //  2.1 MB

  // 1) Casts to bf16 (with K padding for the input GEMM).
  cast_pad_bf16_kernel<<<8192, 256, 0, stream>>>(wrec, wrec_bf, HID, HID, HID * HID);
  cast_pad_bf16_kernel<<<4096, 256, 0, stream>>>(win, win_bf, IN_F, IN_PAD, HID * IN_PAD);
  cast_pad_bf16_kernel<<<4096, 256, 0, stream>>>(x, x_bf, IN_F, IN_PAD,
                                                 T_STEPS * BATCH * IN_PAD);
  // 2) Zero state: vmem,isyn,zsum (f32) + z0 (bf16) are contiguous.
  zero_u32_kernel<<<4096, 256, 0, stream>>>((unsigned int*)vmem,
                                            BATCH * HID * 3 + BATCH * HID / 2);

  // 3) Hoisted input projection for all timesteps: one big WMMA GEMM.
  gemm_xproj_kernel<<<dim3((T_STEPS * BATCH) / 64, HID / 64), 128, 0, stream>>>(
      x_bf, win_bf, xproj);

  // 4) Sequential LIF steps, fused recurrent GEMM + pointwise. Ping-pong z.
  for (int t = 0; t < T_STEPS; ++t) {
    const __hip_bfloat16* zp = (t & 1) ? z1 : z0;
    __hip_bfloat16* zc       = (t & 1) ? z0 : z1;
    lif_step_kernel<<<dim3(BATCH / 64, HID / 64), 128, 0, stream>>>(
        zp, zc, wrec_bf, xproj + (size_t)t * BATCH * HID, vmem, isyn, zsum);
  }

  // 5) Hoisted readout: out = (zsum @ Wro^T)/T + bias.
  readout_kernel<<<BATCH, 256, 0, stream>>>(zsum, wro, bias, out);
}
